// LocalMHSA2D_66434554135185
// MI455X (gfx1250) — matrix operand order repaired and checked by reading, not compiled, not run on hardware
//
#include <hip/hip_runtime.h>

// ---- problem constants -----------------------------------------------------
static constexpr int BATCH  = 8;
static constexpr int CH     = 256;
static constexpr int IMH    = 224;
static constexpr int IMW    = 224;
static constexpr int WSZ    = 8;
static constexpr int NWH    = 28;   // 224 / 8
static constexpr int NWW    = 28;
static constexpr int SEQ    = 64;   // tokens per window
static constexpr int HEADS  = 8;
static constexpr int DH     = 32;   // head dim

// padded LDS row strides (in elements) to avoid bank conflicts
static constexpr int XP = 264;      // X / O tile, f16, 64 rows
static constexpr int QP = 776;      // QKV tile,  f16, 64 rows  (also f32 staging/Y)
static constexpr int AP = 72;       // per-wave attn scratch, f16, 16 rows
static constexpr int YP = 260;      // Y bounce, f32, 64 rows

typedef _Float16 v16h __attribute__((ext_vector_type(16)));
typedef _Float16 v8h  __attribute__((ext_vector_type(8)));
typedef float    v8f  __attribute__((ext_vector_type(8)));
typedef float    v4f  __attribute__((ext_vector_type(4)));
typedef unsigned int u32x4 __attribute__((ext_vector_type(4)));
typedef int          i32x4 __attribute__((ext_vector_type(4)));
typedef int          i32x8 __attribute__((ext_vector_type(8)));

__device__ __forceinline__ v8f wmma16(v16h a, v16h b, v8f c) {
  // D = A(16x32 f16) * B(32x16 f16) + C(16x16 f32)
  return __builtin_amdgcn_wmma_f32_16x16x32_f16(false, a, false, b,
                                                (short)0, c, false, false);
}

// assemble a 16-half fragment from two aligned 8-half LDS chunks
__device__ __forceinline__ v16h ld2x8(const _Float16* p0, const _Float16* p1) {
  union { v16h v; v8h h[2]; } u;
  u.h[0] = *(const v8h*)p0;
  u.h[1] = *(const v8h*)p1;
  return u.v;
}

// load 16 consecutive fp32 weights from global (L2-resident), convert to f16
__device__ __forceinline__ v16h ldw16(const float* p) {
  union { v16h v; _Float16 e[16]; } u;
#pragma unroll
  for (int i = 0; i < 4; ++i) {
    v4f f = *(const v4f*)(p + 4 * i);
    u.e[4*i+0] = (_Float16)f.x;
    u.e[4*i+1] = (_Float16)f.y;
    u.e[4*i+2] = (_Float16)f.z;
    u.e[4*i+3] = (_Float16)f.w;
  }
  return u.v;
}

// low 32 bits of a generic pointer to __shared__ == LDS byte offset
__device__ __forceinline__ unsigned lds_byte_offset(const void* p) {
  return (unsigned)(unsigned long long)p;
}

extern "C" __global__ __launch_bounds__(256, 1)
void lmhsa_win_kernel(const float* __restrict__ x,
                      const float* __restrict__ w_in,
                      const float* __restrict__ b_in,
                      const float* __restrict__ w_out,
                      const float* __restrict__ b_out,
                      float* __restrict__ y)
{
  __shared__ __align__(16) _Float16 lds_x[SEQ * XP];          // X, later O (f16)
  __shared__ __align__(16) _Float16 lds_qkv[SEQ * QP];        // f32 staging -> QKV f16 -> Y f32
  __shared__ __align__(16) _Float16 lds_attn[HEADS * 16 * AP];// per-wave softmax scratch

  const int tid  = threadIdx.x;
  const int lane = tid & 31;
  const int wv   = tid >> 5;      // wave id == head id
  const int g    = lane >> 4;     // half-wave group
  const int lm   = lane & 15;

  const int n  = blockIdx.x;
  const int b  = n / (NWH * NWW);
  const int r  = n % (NWH * NWW);
  const int h0 = (r / NWW) * WSZ;
  const int w0 = (r % NWW) * WSZ;

  // warm L2 for the (shared, resident) weight matrices
  __builtin_prefetch(w_in + (size_t)tid * 768, 0, 1);
  __builtin_prefetch(w_out + (size_t)tid * 256, 0, 1);

  // ---- Stage 0a: TDM async DMA of the whole window into LDS ----
  // tile (W=8, H=8, C=256) of f32 tensor x[b] with dims (224, 224, 256),
  // strides: dim0(H step)=224, dim1(C step)=50176. Lands [c][h][w] contiguous.
  if (wv == 0) {
    const unsigned lds_off = lds_byte_offset(&lds_qkv[0]);
    const unsigned long long ga =
        (unsigned long long)(x + ((size_t)b * CH * IMH * IMW +
                                  (size_t)h0 * IMW + w0));
    u32x4 g0;
    g0.x = 1u;                                        // count=1 (valid D#)
    g0.y = lds_off;                                   // lds_addr
    g0.z = (unsigned)(ga & 0xFFFFFFFFu);              // global_addr[31:0]
    g0.w = (unsigned)((ga >> 32) & 0x01FFFFFFu)       // global_addr[56:32]
           | 0x80000000u;                             // type=2 ("image")

    i32x8 g1;
    g1[0] = 0x00020000;                               // data_size=4B, wg_mask=0
    g1[1] = (int)(224u << 16);                        // tensor_dim0[15:0]
    g1[2] = (int)(224u << 16);                        // tensor_dim1[15:0]
    g1[3] = (int)(8u << 16);                          // tile_dim0 = 8 (W)
    g1[4] = (int)((256u << 16) | 8u);                 // tile_dim2=256 (C) | tile_dim1=8 (H)
    g1[5] = 224;                                      // tensor_dim0_stride = W
    g1[6] = (int)0xC4000000u;                         // tensor_dim1_stride[15:0]=50176 -> bits 223:208
    g1[7] = 0;                                        // tensor_dim1_stride[47:16] = 0

    i32x4 g2;
    g2[0] = 256;                                      // tensor_dim2 = C
    g2[1] = 0; g2[2] = 0; g2[3] = 0;
    i32x4 g3; g3[0] = 0; g3[1] = 0; g3[2] = 0; g3[3] = 0;
    i32x8 g4;                                         // extended/unused D# words
    g4[0] = 0; g4[1] = 0; g4[2] = 0; g4[3] = 0;
    g4[4] = 0; g4[5] = 0; g4[6] = 0; g4[7] = 0;

    __builtin_amdgcn_tensor_load_to_lds(g0, g1, g2, g3, g4, 0);
    __builtin_amdgcn_s_wait_tensorcnt(0);
  }
  __syncthreads();

  // ---- Stage 0b: LDS transpose + f32->f16: staging [c][p] -> token-major X
  {
    const float* stg = (const float*)lds_qkv;   // [c][64] contiguous f32
    const int c = tid;                          // one channel per thread
#pragma unroll
    for (int it = 0; it < 16; ++it) {
      const int pi = ((it + (tid & 15)) & 15) * 4;    // rotated chunk, fewer conflicts
      const v4f f = *(const v4f*)&stg[c * SEQ + pi];
      lds_x[(pi + 0) * XP + c] = (_Float16)f.x;
      lds_x[(pi + 1) * XP + c] = (_Float16)f.y;
      lds_x[(pi + 2) * XP + c] = (_Float16)f.z;
      lds_x[(pi + 3) * XP + c] = (_Float16)f.w;
    }
  }
  __syncthreads();

  // ---- Stage 1: QKV = X(64x256) @ W_in^T(256x768) + b_in  (wave -> its head's cols)
#pragma unroll 1
  for (int part = 0; part < 3; ++part) {          // 0=Q 1=K 2=V
    const int colbase = part * CH + wv * DH;
    v8f acc[2][4] = {};
#pragma unroll 1
    for (int kk = 0; kk < 8; ++kk) {
      const int kb = kk * 32;
      v16h a[4];
#pragma unroll
      for (int m = 0; m < 4; ++m) {
        const _Float16* pa = &lds_x[(m * 16 + lm) * XP + kb];
        a[m] = ld2x8(pa + 16 * g, pa + 16 * g + 8);
      }
#pragma unroll
      for (int nt = 0; nt < 2; ++nt) {
        const int col = colbase + nt * 16 + lm;
        const v16h bf = ldw16(w_in + (size_t)col * CH + kb + g * 16);
#pragma unroll
        for (int m = 0; m < 4; ++m)
          acc[nt][m] = wmma16(a[m], bf, acc[nt][m]);
      }
    }
#pragma unroll
    for (int nt = 0; nt < 2; ++nt) {
      const int col   = colbase + nt * 16 + lm;
      const float bia = b_in[col];
#pragma unroll
      for (int m = 0; m < 4; ++m)
#pragma unroll
        for (int j = 0; j < 8; ++j)
          lds_qkv[(m * 16 + g * 8 + j) * QP + col] = (_Float16)(acc[nt][m][j] + bia);
    }
  }
  __syncthreads();

  // ---- Stage 2: per-head attention (wave wv == head wv) ----
  {
    const float scale = 0.17677669529663687f;   // 1/sqrt(32)
    const int qoff = wv * DH;
    const int koff = CH + wv * DH;
    const int voff = 2 * CH + wv * DH;
    _Float16* At = &lds_attn[wv * 16 * AP];

    // K^T B-fragments (contiguous in token-major K storage)
    v16h ktf[4];
#pragma unroll
    for (int nn = 0; nn < 4; ++nn) {
      const _Float16* pk = &lds_qkv[(nn * 16 + lm) * QP + koff + g * 16];
      ktf[nn] = ld2x8(pk, pk + 8);
    }
    // V B-fragments (strided gather)
    v16h vfr[2][2];
#pragma unroll
    for (int kt = 0; kt < 2; ++kt)
#pragma unroll
      for (int nt = 0; nt < 2; ++nt) {
        v16h t;
#pragma unroll
        for (int i = 0; i < 16; ++i)
          t[i] = lds_qkv[(kt * 32 + g * 16 + i) * QP + voff + nt * 16 + lm];
        vfr[kt][nt] = t;
      }

#pragma unroll 1
    for (int m = 0; m < 4; ++m) {
      const _Float16* pq = &lds_qkv[(m * 16 + lm) * QP + qoff];
      const v16h qf = ld2x8(pq + 16 * g, pq + 16 * g + 8);
      v8f l[4];
#pragma unroll
      for (int nn = 0; nn < 4; ++nn) {
        v8f z = {};
        l[nn] = wmma16(qf, ktf[nn], z);
      }
      // row softmax: rows = (8g+j), cols spread over 4 tiles x 16 lanes
#pragma unroll
      for (int j = 0; j < 8; ++j) {
        float mx = fmaxf(fmaxf(l[0][j] * scale, l[1][j] * scale),
                         fmaxf(l[2][j] * scale, l[3][j] * scale));
        mx = fmaxf(mx, __shfl_xor(mx, 1, 32));
        mx = fmaxf(mx, __shfl_xor(mx, 2, 32));
        mx = fmaxf(mx, __shfl_xor(mx, 4, 32));
        mx = fmaxf(mx, __shfl_xor(mx, 8, 32));
        float e0 = __expf(l[0][j] * scale - mx);
        float e1 = __expf(l[1][j] * scale - mx);
        float e2 = __expf(l[2][j] * scale - mx);
        float e3 = __expf(l[3][j] * scale - mx);
        float sm = e0 + e1 + e2 + e3;
        sm += __shfl_xor(sm, 1, 32);
        sm += __shfl_xor(sm, 2, 32);
        sm += __shfl_xor(sm, 4, 32);
        sm += __shfl_xor(sm, 8, 32);
        const float inv = __builtin_amdgcn_rcpf(sm);
        At[(g * 8 + j) * AP + 0 * 16 + lm] = (_Float16)(e0 * inv);
        At[(g * 8 + j) * AP + 1 * 16 + lm] = (_Float16)(e1 * inv);
        At[(g * 8 + j) * AP + 2 * 16 + lm] = (_Float16)(e2 * inv);
        At[(g * 8 + j) * AP + 3 * 16 + lm] = (_Float16)(e3 * inv);
      }
      __asm volatile("" ::: "memory");  // keep LDS attn stores before reloads

      // O m-block = attn(16x64) @ V(64x32)
      v16h af[2];
#pragma unroll
      for (int kt = 0; kt < 2; ++kt) {
        const _Float16* pa = &At[lm * AP + kt * 32];
        af[kt] = ld2x8(pa + 16 * g, pa + 16 * g + 8);
      }
      v8f oacc[2] = {};
#pragma unroll
      for (int nt = 0; nt < 2; ++nt)
#pragma unroll
        for (int kt = 0; kt < 2; ++kt)
          oacc[nt] = wmma16(af[kt], vfr[kt][nt], oacc[nt]);
      // O -> lds_x (X no longer needed past the stage-1 barrier)
#pragma unroll
      for (int nt = 0; nt < 2; ++nt)
#pragma unroll
        for (int j = 0; j < 8; ++j)
          lds_x[(m * 16 + g * 8 + j) * XP + wv * DH + nt * 16 + lm] =
              (_Float16)oacc[nt][j];
    }
  }
  __syncthreads();

  // ---- Stage 3: Y = O(64x256) @ W_out^T(256x256) + b_out ----
  {
    const int colbase = wv * DH;
    v8f acc[2][4] = {};
#pragma unroll 1
    for (int kk = 0; kk < 8; ++kk) {
      const int kb = kk * 32;
      v16h a[4];
#pragma unroll
      for (int m = 0; m < 4; ++m) {
        const _Float16* pa = &lds_x[(m * 16 + lm) * XP + kb];
        a[m] = ld2x8(pa + 16 * g, pa + 16 * g + 8);
      }
#pragma unroll
      for (int nt = 0; nt < 2; ++nt) {
        const v16h bf =
            ldw16(w_out + (size_t)(colbase + nt * 16 + lm) * CH + kb + g * 16);
#pragma unroll
        for (int m = 0; m < 4; ++m)
          acc[nt][m] = wmma16(a[m], bf, acc[nt][m]);
      }
    }
    // bounce f32 Y through LDS (qkv region is dead now) for coalesced stores
    float* ylds = (float*)lds_qkv;
#pragma unroll
    for (int nt = 0; nt < 2; ++nt) {
      const int col   = colbase + nt * 16 + lm;
      const float bia = b_out[col];
#pragma unroll
      for (int m = 0; m < 4; ++m)
#pragma unroll
        for (int j = 0; j < 8; ++j)
          ylds[(m * 16 + g * 8 + j) * YP + col] = acc[nt][m][j] + bia;
    }
  }
  __syncthreads();

  // ---- coalesced write-out (float2 along W, 8-float contiguous rows) ----
  {
    const float* ylds = (const float*)lds_qkv;
    for (int i = tid; i < SEQ * CH / 2; i += 256) {
      const int c  = i >> 5;              // channel
      const int p  = (i & 31) * 2;        // even pixel index: same row pair
      const int hh = p >> 3, wc = p & 7;
      const size_t ga = (((size_t)b * CH + c) * IMH + (h0 + hh)) * IMW + (w0 + wc);
      float2 v;
      v.x = ylds[(p + 0) * YP + c];
      v.y = ylds[(p + 1) * YP + c];
      *(float2*)&y[ga] = v;
    }
  }
}

extern "C" void kernel_launch(void* const* d_in, const int* in_sizes, int n_in,
                              void* d_out, int out_size, void* d_ws, size_t ws_size,
                              hipStream_t stream) {
  (void)in_sizes; (void)n_in; (void)out_size; (void)d_ws; (void)ws_size;
  const float* x     = (const float*)d_in[0];
  const float* w_in  = (const float*)d_in[1];
  const float* b_in  = (const float*)d_in[2];
  const float* w_out = (const float*)d_in[3];
  const float* b_out = (const float*)d_in[4];
  float* y = (float*)d_out;

  dim3 grid(BATCH * NWH * NWW);   // 6272 windows, one workgroup each
  dim3 block(256);                // 8 waves == 8 heads
  lmhsa_win_kernel<<<grid, block, 0, stream>>>(x, w_in, b_in, w_out, b_out, y);
}
